// GCN_student_11003706212774
// MI455X (gfx1250) — matrix operand order repaired and checked by reading, not compiled, not run on hardware
//
#include <hip/hip_runtime.h>
#include <hip/hip_bf16.h>

// ---------------------------------------------------------------------------
// GCN (DGL GraphConv, norm='both') for MI455X / gfx1250.
//   h = relu( in_norm * Agg( (x * out_norm) @ W ) + b )
// Layers: W1(128->128,relu), 2x W2(128->128,relu), W3(128->40,no relu).
// Output: [ h_final (N x 40) , prior (N x 128) ] concatenated, fp32.
//
// Strategy:
//  * One-time CSR (by dst) build: degree histogram -> 1-block scan -> bucket
//    fill. Turns ~820M scatter atomics into atomic-free pull aggregation.
//  * GEMMs via v_wmma_f32_16x16x32_bf16 (contiguous k = 16*half + j fragments),
//    out_norm scale fused into A-fragment load, B slice held in registers.
//  * Aggregation: wave32 per node, float4 per lane, edge ids via __shfl,
//    next-row prefetch; in_norm/bias/relu fused into the store.
// ---------------------------------------------------------------------------

#define NNODES 100000
#define NEDGES 1600000
#define FDIM   128
#define FCLS   40

typedef __attribute__((ext_vector_type(16))) __bf16 v16bf;
typedef __attribute__((ext_vector_type(8)))  float  v8f;

// ---------------------------- graph preprocessing --------------------------

__global__ __launch_bounds__(256) void k_degrees(const int* __restrict__ src,
                                                 const int* __restrict__ dst,
                                                 int* __restrict__ degout,
                                                 int* __restrict__ degin, int e) {
  int i = blockIdx.x * blockDim.x + threadIdx.x;
  if (i < e) {
    atomicAdd(&degout[src[i]], 1);
    atomicAdd(&degin[dst[i]], 1);
  }
}

__global__ __launch_bounds__(256) void k_norms(const int* __restrict__ degout,
                                               const int* __restrict__ degin,
                                               float* __restrict__ onorm,
                                               float* __restrict__ inorm, int n) {
  int i = blockIdx.x * blockDim.x + threadIdx.x;
  if (i < n) {
    onorm[i] = rsqrtf(fmaxf((float)degout[i], 1.0f));
    inorm[i] = rsqrtf(fmaxf((float)degin[i], 1.0f));
  }
}

// Single-workgroup chunked exclusive scan over N=100K (one-time cost).
__global__ __launch_bounds__(1024) void k_scan(const int* __restrict__ deg,
                                               int* __restrict__ offs, int n) {
  __shared__ int smem[1024];
  __shared__ int carry;
  const int tid = threadIdx.x;
  if (tid == 0) carry = 0;
  __syncthreads();
  for (int base = 0; base < n; base += 1024) {
    int i = base + tid;
    int v = (i < n) ? deg[i] : 0;
    smem[tid] = v;
    __syncthreads();
    for (int off = 1; off < 1024; off <<= 1) {
      int t = (tid >= off) ? smem[tid - off] : 0;
      __syncthreads();
      smem[tid] += t;
      __syncthreads();
    }
    if (i < n) offs[i] = carry + smem[tid] - v;   // exclusive
    __syncthreads();
    if (tid == 1023) carry += smem[1023];
    __syncthreads();
  }
  if (tid == 0) offs[n] = carry;                  // == E
}

__global__ __launch_bounds__(256) void k_fill(const int* __restrict__ src,
                                              const int* __restrict__ dst,
                                              const int* __restrict__ offs,
                                              int* __restrict__ cursor,
                                              int* __restrict__ col, int e) {
  int i = blockIdx.x * blockDim.x + threadIdx.x;
  if (i < e) {
    int d = dst[i];
    int pos = atomicAdd(&cursor[d], 1);
    col[offs[d] + pos] = src[i];
  }
}

// ------------------------------- dense GEMM --------------------------------
// Y[N x OUTW] = (diag(onorm) * X[N x 128]) @ W[128 x OUTW], bf16 WMMA.
// Block: 16-row m-tile per iteration (grid-stride); wave w owns cols 16w..16w+15.
template <int OUTW>
__global__ __launch_bounds__(256) void k_gemm(const float* __restrict__ X,
                                              const float* __restrict__ onorm,
                                              const float* __restrict__ Wm,
                                              float* __restrict__ Y, int mtiles) {
  const int lane = threadIdx.x & 31;
  const int wave = threadIdx.x >> 5;
  const int half = lane >> 4;          // 0: lanes 0-15, 1: lanes 16-31
  const int lmod = lane & 15;
  const int n0   = wave * 16;
  const int ncol = n0 + lmod;          // output column this lane holds in B/C/D

  // Preload B fragments for the whole K=128 (4 steps of K=32).
  // Contiguous k order used here (B 32x16): lane = N, element j -> K = 16*half + j.
  v16bf bfr[4];
#pragma unroll
  for (int s = 0; s < 4; ++s) {
#pragma unroll
    for (int j = 0; j < 16; ++j) {
      int k = s * 32 + half * 16 + j;
      float wv = (ncol < OUTW) ? Wm[k * OUTW + ncol] : 0.0f;
      bfr[s][j] = (__bf16)wv;
    }
  }

  for (int mt = blockIdx.x; mt < mtiles; mt += gridDim.x) {
    const int row = mt * 16 + lmod;    // A: lane = M
    const float sc = onorm[row];
    const float* xr = X + (size_t)row * FDIM;
    v8f c = {};
#pragma unroll
    for (int s = 0; s < 4; ++s) {
      // A 16x32 fragment (matches B): elem j -> K = 16*half + j.
      const int kb = s * 32 + half * 16;
      float4 x0 = *(const float4*)(xr + kb);
      float4 x1 = *(const float4*)(xr + kb + 4);
      float4 x2 = *(const float4*)(xr + kb + 8);
      float4 x3 = *(const float4*)(xr + kb + 12);
      float av[16] = {x0.x, x0.y, x0.z, x0.w, x1.x, x1.y, x1.z, x1.w,
                      x2.x, x2.y, x2.z, x2.w, x3.x, x3.y, x3.z, x3.w};
      v16bf a;
#pragma unroll
      for (int j = 0; j < 16; ++j) a[j] = (__bf16)(av[j] * sc);
      c = __builtin_amdgcn_wmma_f32_16x16x32_bf16(
              /*neg_a=*/false, a, /*neg_b=*/false, bfr[s],
              /*c_mod=*/(short)0, c, /*reuse_a=*/false, /*reuse_b=*/false);
    }
    // C/D layout: VGPR r -> M = r + 8*half, N = ncol.
    float* yr = Y + (size_t)(mt * 16) * OUTW;
    if (ncol < OUTW) {
#pragma unroll
      for (int r = 0; r < 8; ++r) {
        int m = r + half * 8;
        yr[(size_t)m * OUTW + ncol] = c[r];
      }
    }
  }
}

// --------------------------- pull aggregation ------------------------------
// H[d] = act( in_norm[d] * sum_{s in nbr(d)} T[s] + b ), wave32 per node.

__global__ __launch_bounds__(256) void k_agg128(const float* __restrict__ T,
                                                const int* __restrict__ offs,
                                                const int* __restrict__ col,
                                                const float* __restrict__ inorm,
                                                const float* __restrict__ bias,
                                                float* __restrict__ H, int n,
                                                int relu) {
  const int node = blockIdx.x * (blockDim.x >> 5) + (threadIdx.x >> 5);
  if (node >= n) return;
  const int lane = threadIdx.x & 31;
  const int beg = offs[node], end = offs[node + 1];
  float4 acc = {0.f, 0.f, 0.f, 0.f};
  for (int j0 = beg; j0 < end; j0 += 32) {
    int mye = (j0 + lane < end) ? col[j0 + lane] : 0;
    int cnt = min(32, end - j0);
    for (int i = 0; i < cnt; ++i) {
      int s = __shfl(mye, i, 32);
      const float* rp = T + (size_t)s * FDIM + lane * 4;
      if (i + 1 < cnt) {
        int sn = __shfl(mye, i + 1, 32);
        __builtin_prefetch(T + (size_t)sn * FDIM + lane * 4, 0, 0);
      }
      float4 v = *(const float4*)rp;
      acc.x += v.x; acc.y += v.y; acc.z += v.z; acc.w += v.w;
    }
  }
  const float nin = inorm[node];
  const float4 b = *(const float4*)(bias + lane * 4);
  float4 r;
  r.x = acc.x * nin + b.x; r.y = acc.y * nin + b.y;
  r.z = acc.z * nin + b.z; r.w = acc.w * nin + b.w;
  if (relu) {
    r.x = fmaxf(r.x, 0.f); r.y = fmaxf(r.y, 0.f);
    r.z = fmaxf(r.z, 0.f); r.w = fmaxf(r.w, 0.f);
  }
  *(float4*)(H + (size_t)node * FDIM + lane * 4) = r;
}

__global__ __launch_bounds__(256) void k_agg40(const float* __restrict__ T,
                                               const int* __restrict__ offs,
                                               const int* __restrict__ col,
                                               const float* __restrict__ inorm,
                                               const float* __restrict__ bias,
                                               float* __restrict__ H, int n) {
  const int node = blockIdx.x * (blockDim.x >> 5) + (threadIdx.x >> 5);
  if (node >= n) return;
  const int lane = threadIdx.x & 31;
  const int beg = offs[node], end = offs[node + 1];
  float a0 = 0.f, a1 = 0.f;   // features lane and 32+lane (lane<8)
  for (int j0 = beg; j0 < end; j0 += 32) {
    int mye = (j0 + lane < end) ? col[j0 + lane] : 0;
    int cnt = min(32, end - j0);
    for (int i = 0; i < cnt; ++i) {
      int s = __shfl(mye, i, 32);
      const float* rp = T + (size_t)s * FCLS;
      a0 += rp[lane];
      if (lane < 8) a1 += rp[32 + lane];
    }
  }
  const float nin = inorm[node];
  float* hr = H + (size_t)node * FCLS;
  hr[lane] = a0 * nin + bias[lane];
  if (lane < 8) hr[32 + lane] = a1 * nin + bias[32 + lane];
}

// --------------------------------- launch ----------------------------------

extern "C" void kernel_launch(void* const* d_in, const int* in_sizes, int n_in,
                              void* d_out, int out_size, void* d_ws, size_t ws_size,
                              hipStream_t stream) {
  (void)in_sizes; (void)n_in; (void)out_size; (void)ws_size;
  const float* features = (const float*)d_in[0];
  const int*   src      = (const int*)d_in[1];
  const int*   dst      = (const int*)d_in[2];
  const float* W1 = (const float*)d_in[3];
  const float* b1 = (const float*)d_in[4];
  const float* W2 = (const float*)d_in[5];
  const float* b2 = (const float*)d_in[6];
  const float* W3 = (const float*)d_in[7];
  const float* b3 = (const float*)d_in[8];

  const int N = NNODES, E = NEDGES;

  // Workspace carve (~111 MB).
  char* p = (char*)d_ws;
  float* tmp   = (float*)p; p += (size_t)N * FDIM * sizeof(float);
  float* h     = (float*)p; p += (size_t)N * FDIM * sizeof(float);
  float* onorm = (float*)p; p += (size_t)N * sizeof(float);
  float* inorm = (float*)p; p += (size_t)N * sizeof(float);
  int* offs    = (int*)p;   p += (size_t)(N + 1) * sizeof(int);
  int* cursor  = (int*)p;   p += (size_t)N * sizeof(int);
  int* colbuf  = (int*)p;   p += (size_t)E * sizeof(int);
  int* degout  = (int*)p;   p += (size_t)N * sizeof(int);
  int* degin   = (int*)p;   p += (size_t)N * sizeof(int);

  float* out_final = (float*)d_out;                       // N x 40
  float* prior     = (float*)d_out + (size_t)N * FCLS;    // N x 128

  const int mtiles = N / 16;                              // 6250 exact
  const int eblk = (E + 255) / 256;
  const int nblk = (N + 255) / 256;
  const int ablk = (N + 7) / 8;                           // 8 waves/block

  // --- graph preprocessing (rebuilt deterministically every call) ---
  hipMemsetAsync(degout, 0, (size_t)N * sizeof(int), stream);
  hipMemsetAsync(degin,  0, (size_t)N * sizeof(int), stream);
  hipMemsetAsync(cursor, 0, (size_t)N * sizeof(int), stream);
  k_degrees<<<eblk, 256, 0, stream>>>(src, dst, degout, degin, E);
  k_norms<<<nblk, 256, 0, stream>>>(degout, degin, onorm, inorm, N);
  k_scan<<<1, 1024, 0, stream>>>(degin, offs, N);
  k_fill<<<eblk, 256, 0, stream>>>(src, dst, offs, cursor, colbuf, E);

  // --- layer 1: features -> h ---
  k_gemm<FDIM><<<768, 256, 0, stream>>>(features, onorm, W1, tmp, mtiles);
  k_agg128<<<ablk, 256, 0, stream>>>(tmp, offs, colbuf, inorm, b1, h, N, 1);

  // --- shared middle layer, repeat 1: h -> h ---
  k_gemm<FDIM><<<768, 256, 0, stream>>>(h, onorm, W2, tmp, mtiles);
  k_agg128<<<ablk, 256, 0, stream>>>(tmp, offs, colbuf, inorm, b2, h, N, 1);

  // --- shared middle layer, repeat 2: h -> prior (second output) ---
  k_gemm<FDIM><<<768, 256, 0, stream>>>(h, onorm, W2, tmp, mtiles);
  k_agg128<<<ablk, 256, 0, stream>>>(tmp, offs, colbuf, inorm, b2, prior, N, 1);

  // --- classifier layer: prior -> out_final (no relu) ---
  k_gemm<FCLS><<<768, 96, 0, stream>>>(prior, onorm, W3, tmp, mtiles);
  k_agg40<<<ablk, 256, 0, stream>>>(tmp, offs, colbuf, inorm, b3, out_final, N);
}
